// TransformerRegressor_74105365725261
// MI455X (gfx1250) — matrix operand order repaired and checked by reading, not compiled, not run on hardware
//
#include <hip/hip_runtime.h>

typedef __attribute__((ext_vector_type(16))) _Float16 v16h;
typedef __attribute__((ext_vector_type(8)))  _Float16 v8h;
typedef __attribute__((ext_vector_type(4)))  _Float16 v4h;
typedef __attribute__((ext_vector_type(8)))  float    v8f;
typedef __attribute__((ext_vector_type(4)))  float    v4f;

namespace {
constexpr int B_ = 4, S_ = 1024, FIN_ = 3, FM_ = 5;
constexpr int L_ = 4, D_ = 256, H_ = 8, FF_ = 1024, OUT_ = 20, DH_ = 32;
constexpr float LN_EPS = 1e-5f;
}

union V8F { v8f v; float f[8]; };

__device__ inline v16h cat8(v8h lo, v8h hi){
  union { v16h w; v8h h[2]; } u; u.h[0] = lo; u.h[1] = hi; return u.w;
}

__device__ inline v8f wmma_f32_f16(v16h a, v16h b, v8f c){
  // D = A(16x32 f16) * B(32x16 f16) + C(16x16 f32)
  return __builtin_amdgcn_wmma_f32_16x16x32_f16(false, a, false, b, (short)0, c,
                                                false, false);
}

// A fragment from a 16x32 f16 tile in LDS (row-major, row stride 32 halves).
// CDNA5 16-bit A layout: lanes 0-15 row m hold K{0..7,16..23}; lanes 16-31 K{8..15,24..31}.
__device__ inline v16h load_a_frag(const _Float16* tile, int lane){
  int m  = lane & 15;
  int hi = lane >> 4;
  const v8h* row = (const v8h*)(tile + m * 32);
  return cat8(row[hi], row[2 + hi]);
}

// B fragment: tile rows are "n" (output column / key), 32 contiguous K halves per row.
// CDNA5 16-bit B layout: lane n (0-15) holds K{0..7,16..23} of column n; lanes 16-31 hold K{8..15,24..31}.
__device__ inline v16h load_b_frag(const _Float16* tile, int nbase, int lane){
  int n  = nbase + (lane & 15);
  int hi = lane >> 4;
  const v8h* row = (const v8h*)(tile + n * 32);
  return cat8(row[hi], row[2 + hi]);
}

// ---------------------------------------------------------------------------
// conversions / padding
// ---------------------------------------------------------------------------
__global__ void __launch_bounds__(256)
f32_to_f16_kernel(const float* __restrict__ src, _Float16* __restrict__ dst, int n){
  int i = blockIdx.x * 256 + threadIdx.x;
  if(i < n) dst[i] = (_Float16)src[i];
}

// w_in [D,3] -> f16 [D,32] zero-padded in K
__global__ void __launch_bounds__(256)
convert_pad_win_kernel(const float* __restrict__ src, _Float16* __restrict__ dst){
  int i = blockIdx.x * 256 + threadIdx.x;   // D_*32 elements
  int r = i >> 5, c = i & 31;
  dst[i] = (c < FIN_) ? (_Float16)src[r * FIN_ + c] : (_Float16)0.f;
}

// input_coord [M,3] -> f32 [M,32] zero-padded (makes every GEMM dense, K%32==0)
__global__ void __launch_bounds__(256)
pad_input_kernel(const float* __restrict__ src, float* __restrict__ dst){
  int i = blockIdx.x * 256 + threadIdx.x;   // M*32 elements
  int r = i >> 5, c = i & 31;
  dst[i] = (c < FIN_) ? src[r * FIN_ + c] : 0.f;
}

// ---------------------------------------------------------------------------
// geometric distance mask, packed into 32-key ballot words (wave32: lane == k%32)
// word index = ((b*S + q)*S + k)/32 ; bit = k%32 ; 1 = blocked
// ---------------------------------------------------------------------------
__global__ void __launch_bounds__(256)
mask_kernel(const float* __restrict__ pts, const unsigned char* __restrict__ pad,
            unsigned int* __restrict__ blockedw){
  long i = (long)blockIdx.x * 256 + threadIdx.x;   // exactly B*S*S threads
  int  k = (int)(i % S_);
  long t = i / S_;
  int  q = (int)(t % S_);
  int  b = (int)(t / S_);
  bool pq  = pad[b * S_ + q] != 0;
  bool pk  = pad[b * S_ + k] != 0;
  bool npm = !(pq || pk || (q == k));
  const float big = 1e6f;
  const float PI  = 3.14159265358979323846f;
  const float* P1 = pts + ((long)b * S_ + q) * FM_;   // "point1" = query
  const float* P2 = pts + ((long)b * S_ + k) * FM_;   // "point2" = key
  float z1 = P1[0], r1 = P1[1], p1 = P1[2], e1 = P1[3];
  float z2 = P2[0], r2 = P2[1], p2 = P2[2], e2 = P2[3];
  float rd = npm ? (r2 - r1) : big;
  float z0 = (rd != 0.f && npm) ? fabsf(z1 - r1 * (z2 - z1) / rd) : big;
  float pd = npm ? fabsf(p2 - p1) : big;
  pd = (pd > PI && npm) ? (2.f * PI - pd) : pd;
  float ratio = npm ? pd / (fabsf(rd) + 1e-8f) : big;
  float ang   = npm ? sqrtf((e2 - e1) * (e2 - e1) + pd * pd) : big;
  bool dm  = npm && (z0 > 197.4f) && (ratio > 0.001825f) && (ang > 1.797f);
  bool blk = dm || pk;
  unsigned int w = __builtin_amdgcn_ballot_w32(blk);
  if((threadIdx.x & 31) == 0) blockedw[i >> 5] = w;
}

// ---------------------------------------------------------------------------
// GEMM: C[M,N] = act(A[M,K(lda)] * W[N,K(ldw)]^T + bias[N])
// block tile 128x64, 8 waves; wave w owns rows w*16..w*16+15 and all 64 cols:
// per K-step: 1 A-frag + 4 B-frags -> 4 WMMAs. Dense staging (K%32==0, lda%4==0).
// ---------------------------------------------------------------------------
__global__ void __launch_bounds__(256)
gemm_bias_kernel(const float* __restrict__ Ag, int lda,
                 const _Float16* __restrict__ Wg, int ldw,
                 const float* __restrict__ bias, float* __restrict__ Cg,
                 int M, int N, int K, int relu){
  __shared__ _Float16 sA[128 * 32];
  __shared__ _Float16 sW[64 * 32];
  const int tid  = threadIdx.x;
  const int lane = tid & 31;
  const int wave = tid >> 5;
  const int m0   = blockIdx.x * 128;
  const int n0   = blockIdx.y * 64;
  // W staging map: 64 rows x 4 chunks of 8 halves
  const int wr = tid >> 2, wc = tid & 3;
  const int wgn = n0 + wr;
  v8f acc0 = {}, acc1 = {}, acc2 = {}, acc3 = {};
  for(int k0 = 0; k0 < K; k0 += 32){
    // stage A tile 128x32: 1024 float4 chunks, 4 per thread (global_load_b128)
#pragma unroll
    for(int t = 0; t < 4; t++){
      int chunk = tid + t * 256;
      int r = chunk >> 3, c = chunk & 7;
      v4f a4 = *(const v4f*)(Ag + (long)(m0 + r) * lda + k0 + c * 4);
      *(v4h*)(sA + r * 32 + c * 4) = __builtin_convertvector(a4, v4h);
    }
    // stage W tile 64x32 halves (one b128 load per thread)
    {
      v8h val = {};
      if(wgn < N) val = *(const v8h*)(Wg + (long)wgn * ldw + k0 + wc * 8);
      *(v8h*)(sW + wr * 32 + wc * 8) = val;
    }
    // prefetch next K tile (global_prefetch_b8)
    if(k0 + 32 < K){
      __builtin_prefetch(Ag + (long)(m0 + (tid >> 3)) * lda + k0 + 32 + (tid & 7) * 4, 0, 1);
      if(wgn < N) __builtin_prefetch(Wg + (long)wgn * ldw + k0 + 32 + wc * 8, 0, 1);
    }
    __syncthreads();
    v16h a  = load_a_frag(sA + wave * 16 * 32, lane);
    v16h b0 = load_b_frag(sW, 0,  lane);
    v16h b1 = load_b_frag(sW, 16, lane);
    v16h b2 = load_b_frag(sW, 32, lane);
    v16h b3 = load_b_frag(sW, 48, lane);
    acc0 = wmma_f32_f16(a, b0, acc0);
    acc1 = wmma_f32_f16(a, b1, acc1);
    acc2 = wmma_f32_f16(a, b2, acc2);
    acc3 = wmma_f32_f16(a, b3, acc3);
    __syncthreads();
  }
  const int hi = lane >> 4, nn = lane & 15;
  const int mbase = m0 + wave * 16 + hi * 8;   // C layout: VGPR r -> row r (+8 for hi half)
  v8f accs[4] = {acc0, acc1, acc2, acc3};
#pragma unroll
  for(int j = 0; j < 4; j++){
    int nc = n0 + j * 16 + nn;
    if(nc < N){
      float bv = bias[nc];
      V8F a0; a0.v = accs[j];
#pragma unroll
      for(int r = 0; r < 8; r++){
        float v = a0.f[r] + bv;
        if(relu) v = fmaxf(v, 0.f);
        Cg[(long)(mbase + r) * N + nc] = v;
      }
    }
  }
}

// ---------------------------------------------------------------------------
// flash attention: grid (S/128, H, B); 8 waves, each owns a 16-row q tile.
// qkv layout [B*S, 768]: Q = cols 0..255, K = 256..511, V = 512..767; head h = dims h*32..+31
// ---------------------------------------------------------------------------
__global__ void __launch_bounds__(256)
attention_kernel(const float* __restrict__ qkv, const unsigned int* __restrict__ blockedw,
                 float* __restrict__ outx){
  __shared__ _Float16 sK[32 * 32];        // [key][d]
  __shared__ _Float16 sV[32 * 32];        // transposed: [d][key]
  __shared__ _Float16 sP[8][16 * 32];     // per-wave staging (Q fragment, then P)
  const int b = blockIdx.z, h = blockIdx.y;
  const int tid = threadIdx.x, lane = tid & 31, wave = tid >> 5;
  const int q0 = blockIdx.x * 128 + wave * 16;
  const int hi = lane >> 4, nn = lane & 15;
  const long rstr = 3 * D_;
  const float* Qb = qkv + (long)b * S_ * rstr + h * DH_;
  const float* Kb = Qb + D_;
  const float* Vb = Qb + 2 * D_;
  // stage this wave's Q tile (16x32) vectorized, keep A fragment in registers
  {
    _Float16* st = sP[wave];
#pragma unroll
    for(int t = 0; t < 4; t++){
      int chunk = lane + t * 32;          // 128 chunks: 16 rows x 8 float4s
      int r = chunk >> 3, c = chunk & 7;
      v4f q4 = *(const v4f*)(Qb + (long)(q0 + r) * rstr + c * 4);
      *(v4h*)(st + r * 32 + c * 4) = __builtin_convertvector(q4, v4h);
    }
  }
  asm volatile("s_wait_dscnt 0" ::: "memory");
  const v16h aq = load_a_frag(sP[wave], lane);
  V8F o0, o1;
  { v8f z = {}; o0.v = z; o1.v = z; }
  float rm[8], rsum[8];
#pragma unroll
  for(int r = 0; r < 8; r++){ rm[r] = -3.0e38f; rsum[r] = 0.f; }
  const float scale = 0.17677669529663687f;   // 1/sqrt(32)
  const unsigned int* mw = blockedw + ((long)b * S_ + q0) * (S_ / 32);
  const int kvr = tid >> 3, kvc = tid & 7;    // K/V staging: 32 rows x 8 float4s
  for(int kt = 0; kt < S_ / 32; kt++){
    const int k0 = kt * 32;
    __syncthreads();
    {
      v4f k4 = *(const v4f*)(Kb + (long)(k0 + kvr) * rstr + kvc * 4);
      v4f vv = *(const v4f*)(Vb + (long)(k0 + kvr) * rstr + kvc * 4);
      *(v4h*)(sK + kvr * 32 + kvc * 4) = __builtin_convertvector(k4, v4h);
      v4h vh = __builtin_convertvector(vv, v4h);
      sV[(kvc * 4 + 0) * 32 + kvr] = vh.x;    // transpose V: [d][key]
      sV[(kvc * 4 + 1) * 32 + kvr] = vh.y;
      sV[(kvc * 4 + 2) * 32 + kvr] = vh.z;
      sV[(kvc * 4 + 3) * 32 + kvr] = vh.w;
    }
    __syncthreads();
    V8F s0, s1;
    { v8f z = {};
      s0.v = wmma_f32_f16(aq, load_b_frag(sK, 0, lane), z);
      s1.v = wmma_f32_f16(aq, load_b_frag(sK, 16, lane), z); }
#pragma unroll
    for(int r = 0; r < 8; r++){
      int m = r + hi * 8;
      unsigned int w = mw[(long)m * (S_ / 32) + kt];
      float c0 = s0.f[r] * scale;
      float c1 = s1.f[r] * scale;
      if((w >> nn) & 1u)        c0 = -1e9f;
      if((w >> (16 + nn)) & 1u) c1 = -1e9f;
      float tmax = fmaxf(c0, c1);
      tmax = fmaxf(tmax, __shfl_xor(tmax, 1, 16));
      tmax = fmaxf(tmax, __shfl_xor(tmax, 2, 16));
      tmax = fmaxf(tmax, __shfl_xor(tmax, 4, 16));
      tmax = fmaxf(tmax, __shfl_xor(tmax, 8, 16));
      float mnew = fmaxf(rm[r], tmax);
      float corr = __expf(rm[r] - mnew);
      float p0 = __expf(c0 - mnew);
      float p1 = __expf(c1 - mnew);
      float ps = p0 + p1;
      ps += __shfl_xor(ps, 1, 16);
      ps += __shfl_xor(ps, 2, 16);
      ps += __shfl_xor(ps, 4, 16);
      ps += __shfl_xor(ps, 8, 16);
      rsum[r] = rsum[r] * corr + ps;
      rm[r] = mnew;
      o0.f[r] *= corr;
      o1.f[r] *= corr;
      s0.f[r] = p0;
      s1.f[r] = p1;
    }
    // transpose P (C-layout accs) into an A fragment through per-wave LDS
    _Float16* st = sP[wave];
#pragma unroll
    for(int r = 0; r < 8; r++){
      int m = r + hi * 8;
      st[m * 32 + nn]      = (_Float16)s0.f[r];
      st[m * 32 + 16 + nn] = (_Float16)s1.f[r];
    }
    asm volatile("s_wait_dscnt 0" ::: "memory");
    v16h ap = load_a_frag(sP[wave], lane);
    o0.v = wmma_f32_f16(ap, load_b_frag(sV, 0, lane), o0.v);   // dims 0..15
    o1.v = wmma_f32_f16(ap, load_b_frag(sV, 16, lane), o1.v);  // dims 16..31
  }
  float* orow = outx + ((long)b * S_ + q0) * D_ + h * DH_;
#pragma unroll
  for(int r = 0; r < 8; r++){
    int m = r + hi * 8;
    float inv = 1.f / rsum[r];
    orow[(long)m * D_ + nn]      = o0.f[r] * inv;
    orow[(long)m * D_ + 16 + nn] = o1.f[r] * inv;
  }
}

// ---------------------------------------------------------------------------
// out = LayerNorm(x + res) * g + b ; one block per row, in-place safe
// ---------------------------------------------------------------------------
__global__ void __launch_bounds__(256)
add_ln_kernel(const float* __restrict__ x, const float* __restrict__ res,
              const float* __restrict__ g, const float* __restrict__ bt,
              float* __restrict__ out){
  __shared__ float red[8];
  const int row = blockIdx.x, t = threadIdx.x;
  const long idx = (long)row * D_ + t;
  float v = x[idx] + res[idx];
  float s = v;
  for(int o = 16; o > 0; o >>= 1) s += __shfl_xor(s, o, 32);
  if((t & 31) == 0) red[t >> 5] = s;
  __syncthreads();
  float tot = 0.f;
  for(int i = 0; i < 8; i++) tot += red[i];
  const float mean = tot * (1.f / D_);
  const float d = v - mean;
  __syncthreads();
  float s2 = d * d;
  for(int o = 16; o > 0; o >>= 1) s2 += __shfl_xor(s2, o, 32);
  if((t & 31) == 0) red[t >> 5] = s2;
  __syncthreads();
  float tv = 0.f;
  for(int i = 0; i < 8; i++) tv += red[i];
  const float var = tv * (1.f / D_);
  out[idx] = d * rsqrtf(var + LN_EPS) * g[t] + bt[t];
}

// ---------------------------------------------------------------------------
extern "C" void kernel_launch(void* const* d_in, const int* in_sizes, int n_in,
                              void* d_out, int out_size, void* d_ws, size_t ws_size,
                              hipStream_t stream){
  (void)in_sizes; (void)n_in; (void)out_size; (void)ws_size;
  const float* input_coord    = (const float*)d_in[0];
  const float* input_for_mask = (const float*)d_in[1];
  const unsigned char* padm   = (const unsigned char*)d_in[2];  // numpy bool = 1 byte
  const float* w_in  = (const float*)d_in[3];
  const float* b_in  = (const float*)d_in[4];
  const float* w_qkv = (const float*)d_in[5];
  const float* b_qkv = (const float*)d_in[6];
  const float* w_o   = (const float*)d_in[7];
  const float* b_o   = (const float*)d_in[8];
  const float* w_ff1 = (const float*)d_in[9];
  const float* b_ff1 = (const float*)d_in[10];
  const float* w_ff2 = (const float*)d_in[11];
  const float* b_ff2 = (const float*)d_in[12];
  const float* ln1_g = (const float*)d_in[13];
  const float* ln1_b = (const float*)d_in[14];
  const float* ln2_g = (const float*)d_in[15];
  const float* ln2_b = (const float*)d_in[16];
  const float* w_dec = (const float*)d_in[17];
  const float* b_dec = (const float*)d_in[18];
  float* out = (float*)d_out;

  char* ws = (char*)d_ws;
  size_t off = 0;
  auto alloc = [&](size_t bytes) -> void* {
    void* p = ws + off;
    off = (off + bytes + 255) & ~(size_t)255;
    return p;
  };
  unsigned int* maskw = (unsigned int*)alloc((size_t)B_ * S_ * (S_ / 32) * 4);
  _Float16* win_h  = (_Float16*)alloc((size_t)D_ * 32 * 2);
  _Float16* wqkv_h = (_Float16*)alloc((size_t)L_ * 3 * D_ * D_ * 2);
  _Float16* wo_h   = (_Float16*)alloc((size_t)L_ * D_ * D_ * 2);
  _Float16* wff1_h = (_Float16*)alloc((size_t)L_ * FF_ * D_ * 2);
  _Float16* wff2_h = (_Float16*)alloc((size_t)L_ * D_ * FF_ * 2);
  _Float16* wdec_h = (_Float16*)alloc((size_t)OUT_ * D_ * 2);
  float* xpad = (float*)alloc((size_t)B_ * S_ * 32 * 4);
  float* x    = (float*)alloc((size_t)B_ * S_ * D_ * 4);
  float* qkvb = (float*)alloc((size_t)B_ * S_ * 3 * D_ * 4);
  float* attn = (float*)alloc((size_t)B_ * S_ * D_ * 4);
  float* tmp  = (float*)alloc((size_t)B_ * S_ * D_ * 4);
  float* ffb  = (float*)alloc((size_t)B_ * S_ * FF_ * 4);

  const int M = B_ * S_;
  auto cvt = [&](const float* src, _Float16* dst, int n){
    f32_to_f16_kernel<<<(n + 255) / 256, 256, 0, stream>>>(src, dst, n);
  };
  convert_pad_win_kernel<<<(D_ * 32 + 255) / 256, 256, 0, stream>>>(w_in, win_h);
  pad_input_kernel<<<(M * 32 + 255) / 256, 256, 0, stream>>>(input_coord, xpad);
  cvt(w_qkv, wqkv_h, L_ * 3 * D_ * D_);
  cvt(w_o,   wo_h,   L_ * D_ * D_);
  cvt(w_ff1, wff1_h, L_ * FF_ * D_);
  cvt(w_ff2, wff2_h, L_ * D_ * FF_);
  cvt(w_dec, wdec_h, OUT_ * D_);
  mask_kernel<<<(int)(((long)B_ * S_ * S_) / 256), 256, 0, stream>>>(
      input_for_mask, padm, maskw);

  auto gemm = [&](const float* A, int lda, const _Float16* W, int ldw,
                  const float* bias, float* C, int N, int K, int relu){
    dim3 grid(M / 128, (N + 63) / 64);
    gemm_bias_kernel<<<grid, 256, 0, stream>>>(A, lda, W, ldw, bias, C, M, N, K, relu);
  };

  // input projection: padded [M,32] @ [256,32]^T
  gemm(xpad, 32, win_h, 32, b_in, x, D_, 32, 0);
  for(int l = 0; l < L_; l++){
    gemm(x, D_, wqkv_h + (size_t)l * 3 * D_ * D_, D_, b_qkv + l * 3 * D_, qkvb, 3 * D_, D_, 0);
    dim3 gatt(S_ / 128, H_, B_);
    attention_kernel<<<gatt, 256, 0, stream>>>(qkvb, maskw, attn);
    gemm(attn, D_, wo_h + (size_t)l * D_ * D_, D_, b_o + l * D_, tmp, D_, D_, 0);
    add_ln_kernel<<<M, 256, 0, stream>>>(x, tmp, ln1_g + l * D_, ln1_b + l * D_, x);
    gemm(x, D_, wff1_h + (size_t)l * FF_ * D_, D_, b_ff1 + l * FF_, ffb, FF_, D_, 1);
    gemm(ffb, FF_, wff2_h + (size_t)l * D_ * FF_, FF_, b_ff2 + l * D_, tmp, D_, FF_, 0);
    add_ln_kernel<<<M, 256, 0, stream>>>(x, tmp, ln2_g + l * D_, ln2_b + l * D_, x);
  }
  gemm(x, D_, wdec_h, D_, b_dec, out, OUT_, D_, 0);
}
